// WordEmbedding_46402826666034
// MI455X (gfx1250) — matrix operand order repaired and checked by reading, not compiled, not run on hardware
//
#include <hip/hip_runtime.h>
#include <hip/hip_bf16.h>

// ---------------------------------------------------------------------------
// Shapes / offsets
// ---------------------------------------------------------------------------
#define VOCAB   32000
#define DIM     1024
#define BITS    15
#define NTOK    4096            // 4 * 1024 tokens
#define OUT_ID_EMB   ((size_t)0)
#define OUT_BIT_EMB  ((size_t)4194304)          // 4096*1024
#define OUT_LOGIT    ((size_t)8388608)          // + 4096*1024
#define OUT_LOGIT_W  ((size_t)139460608)        // + 4096*32000

// workspace layout (bytes): bf16 hi/lo planes of tensor and weight
#define WS_A_HI  ((size_t)0)
#define WS_A_LO  ((size_t)8388608)              // 4096*1024*2
#define WS_W_HI  ((size_t)16777216)
#define WS_W_LO  ((size_t)82313216)             // + 32000*1024*2
// total ws needed: 147,849,216 bytes (~141 MB)

// LDS tile geometry (rows padded 32 -> 40 elems = 80B stride: conflict-free b128)
#define KSTEP 32
#define KPAD  40

// ---------------------------------------------------------------------------
// WMMA types (CDNA5 / gfx1250, wave32)
// ---------------------------------------------------------------------------
typedef __attribute__((ext_vector_type(16))) __bf16          v16bf;
typedef __attribute__((ext_vector_type(8)))  float           v8f;
typedef __attribute__((ext_vector_type(8)))  unsigned short  u16x8;
typedef __attribute__((ext_vector_type(4)))  int             v4i;

union BF16Frag { v16bf v; u16x8 h[2]; };

struct __attribute__((aligned(8))) us4 { unsigned short x, y, z, w; };

#if __has_builtin(__builtin_amdgcn_global_load_async_to_lds_b128)
#define HAVE_ASYNC_LDS 1
typedef __attribute__((address_space(1))) v4i glb_v4i;   // global-AS 16B chunk
typedef __attribute__((address_space(3))) v4i lds_v4i;   // LDS-AS   16B chunk
#endif

__device__ __forceinline__ unsigned short f2bf_rne(float f) {
    unsigned u = __float_as_uint(f);
    u += 0x7FFFu + ((u >> 16) & 1u);          // round-to-nearest-even
    return (unsigned short)(u >> 16);
}
__device__ __forceinline__ float bf2f(unsigned short s) {
    return __uint_as_float(((unsigned)s) << 16);
}
// two 128-bit loads -> one 16-element bf16 fragment
__device__ __forceinline__ v16bf load_frag(const unsigned short* p0,
                                           const unsigned short* p1) {
    BF16Frag f;
    f.h[0] = *(const u16x8*)p0;
    f.h[1] = *(const u16x8*)p1;
    return f.v;
}
// stage 16B global -> LDS (async on CDNA5, plain copy fallback)
__device__ __forceinline__ void stage16B(const unsigned short* g, unsigned short* l) {
#ifdef HAVE_ASYNC_LDS
    __builtin_amdgcn_global_load_async_to_lds_b128((glb_v4i*)g, (lds_v4i*)l, 0, 0);
#else
    *(u16x8*)l = *(const u16x8*)g;
#endif
}
__device__ __forceinline__ void stage_wait() {
#ifdef HAVE_ASYNC_LDS
    asm volatile("s_wait_asynccnt 0x0" ::: "memory");
#endif
    __syncthreads();
}

// ---------------------------------------------------------------------------
// Kernel 0: split fp32 plane -> bf16 hi plane + bf16 lo plane (x ~= hi + lo)
// ---------------------------------------------------------------------------
__global__ __launch_bounds__(256) void split_planes_kernel(
    const float* __restrict__ src,
    unsigned short* __restrict__ hi,
    unsigned short* __restrict__ lo,
    int n4)
{
    int i = blockIdx.x * 256 + threadIdx.x;
    if (i >= n4) return;
    float4 v = ((const float4*)src)[i];
    us4 h, l;
    h.x = f2bf_rne(v.x); l.x = f2bf_rne(v.x - bf2f(h.x));
    h.y = f2bf_rne(v.y); l.y = f2bf_rne(v.y - bf2f(h.y));
    h.z = f2bf_rne(v.z); l.z = f2bf_rne(v.z - bf2f(h.z));
    h.w = f2bf_rne(v.w); l.w = f2bf_rne(v.w - bf2f(h.w));
    ((us4*)hi)[i] = h;
    ((us4*)lo)[i] = l;
}

// ---------------------------------------------------------------------------
// Kernel 1: fused id_emb gather + bit_emb (bits @ weight_bit)
// ---------------------------------------------------------------------------
__global__ __launch_bounds__(256) void embed_kernel(
    const int* __restrict__ ids,
    const float* __restrict__ weight,
    const float* __restrict__ weight_bit,
    float* __restrict__ id_emb,
    float* __restrict__ bit_emb)
{
    __shared__ float4 wb[BITS * (DIM / 4)];   // 15 * 256 float4 = 60 KB
    const int tid = threadIdx.x;
    const int tok = blockIdx.x;

    for (int i = tid; i < BITS * (DIM / 4); i += 256)
        wb[i] = ((const float4*)weight_bit)[i];

    int id = ids[tok];
    id = id < 0 ? 0 : (id > VOCAB - 1 ? VOCAB - 1 : id);
    float bits[BITS];
#pragma unroll
    for (int k = 0; k < BITS; ++k)
        bits[k] = ((id >> (BITS - 1 - k)) & 1) ? 1.0f : -1.0f;

    __syncthreads();

    float4 w = ((const float4*)(weight + (size_t)id * DIM))[tid];
    ((float4*)(id_emb + (size_t)tok * DIM))[tid] = w;

    float4 acc = {0.f, 0.f, 0.f, 0.f};
#pragma unroll
    for (int k = 0; k < BITS; ++k) {
        float4 wv = wb[k * (DIM / 4) + tid];
        acc.x += bits[k] * wv.x;
        acc.y += bits[k] * wv.y;
        acc.z += bits[k] * wv.z;
        acc.w += bits[k] * wv.w;
    }
    ((float4*)(bit_emb + (size_t)tok * DIM))[tid] = acc;
}

// ---------------------------------------------------------------------------
// Kernel 2: logit[4096,32000] = tensor @ weight^T via bf16 split precision
// on v_wmma_f32_16x16x32_bf16, operands pre-split into bf16 planes in ws.
// Block = 256 threads = 8 waves; block tile 64(M) x 128(N), K-step 32,
// double-buffered LDS staging via async-load-to-LDS (ASYNCcnt).
// Wave (wm 0..3, wn 0..1) computes 16(M) x 64(N) = four 16x16 accumulators.
// ---------------------------------------------------------------------------
__global__ __launch_bounds__(256) void gemm_logit_kernel(
    const unsigned short* __restrict__ a_hi,
    const unsigned short* __restrict__ a_lo,
    const unsigned short* __restrict__ w_hi,
    const unsigned short* __restrict__ w_lo,
    float* __restrict__ logit)
{
    __shared__ unsigned short sA[2][2][64][KPAD];    // [buf][hi/lo][row][k]
    __shared__ unsigned short sB[2][2][128][KPAD];   // [buf][hi/lo][col][k]

    const int tid  = threadIdx.x;
    const int lane = tid & 31;
    const int wave = tid >> 5;
    const int wm   = wave >> 1;          // 0..3  (M strip within block)
    const int wn   = wave & 1;           // 0..1  (N strip within block)
    const int h    = lane >> 4;          // half-wave (0/1)
    const int ml   = lane & 15;

    const int mBase = blockIdx.y * 64;
    const int nBase = blockIdx.x * 128;
    const int n0    = nBase + wn * 64;   // four 16-col subtiles per wave

    // cooperative tile staging: 6 x 16B per thread per K-step
    auto stage = [&](int kt, int buf) {
#pragma unroll
        for (int r = 0; r < 2; ++r) {               // A: 2 planes x 64 rows x 4 chunks
            int q = tid + r * 256;
            int plane = q >> 8, row = (q & 255) >> 2, sub = q & 3;
            const unsigned short* src =
                (plane ? a_lo : a_hi) + (size_t)(mBase + row) * DIM + kt + sub * 8;
            stage16B(src, &sA[buf][plane][row][sub * 8]);
        }
#pragma unroll
        for (int r = 0; r < 4; ++r) {               // B: 2 planes x 128 cols x 4 chunks
            int q = tid + r * 256;
            int plane = q >> 9, row = (q & 511) >> 2, sub = q & 3;
            const unsigned short* src =
                (plane ? w_lo : w_hi) + (size_t)(nBase + row) * DIM + kt + sub * 8;
            stage16B(src, &sB[buf][plane][row][sub * 8]);
        }
    };

    v8f acc[4];
#pragma unroll
    for (int st = 0; st < 4; ++st) acc[st] = (v8f){};

    stage(0, 0);
    stage_wait();

    const int arow = wm * 16 + ml;       // A fragment row for this lane

    for (int kt = 0, it = 0; kt < DIM; kt += KSTEP, ++it) {
        const int buf = it & 1;
        if (kt + KSTEP < DIM)
            stage(kt + KSTEP, buf ^ 1);

        // A fragment (16-bit A 16x32 layout):
        //   e=0..7 -> K = h*8 + e ; e=8..15 -> K = 16 + h*8 + (e-8)
        v16bf ahi = load_frag(&sA[buf][0][arow][h * 8], &sA[buf][0][arow][16 + h * 8]);
        v16bf alo = load_frag(&sA[buf][1][arow][h * 8], &sA[buf][1][arow][16 + h * 8]);

#pragma unroll
        for (int st = 0; st < 4; ++st) {
            // B fragment (B 32x16: lane ml = column, e=0..7 -> K = h*8 + e ; e=8..15 -> K = 16 + h*8 + (e-8))
            const int bcol = wn * 64 + st * 16 + ml;
            v16bf bhi = load_frag(&sB[buf][0][bcol][h * 8], &sB[buf][0][bcol][16 + h * 8]);
            v16bf blo = load_frag(&sB[buf][1][bcol][h * 8], &sB[buf][1][bcol][16 + h * 8]);

            acc[st] = __builtin_amdgcn_wmma_f32_16x16x32_bf16(
                false, ahi, false, bhi, (short)0, acc[st], false, false);
            acc[st] = __builtin_amdgcn_wmma_f32_16x16x32_bf16(
                false, ahi, false, blo, (short)0, acc[st], false, false);
            acc[st] = __builtin_amdgcn_wmma_f32_16x16x32_bf16(
                false, alo, false, bhi, (short)0, acc[st], false, false);
        }

        stage_wait();   // next buffer ready; all waves done reading this buffer
    }

    // store: D VGPR r -> M = mBase + wm*16 + r + 8*h, N = n0 + st*16 + ml
    // non-temporal so the 524MB logit stream doesn't evict L2-resident planes
#pragma unroll
    for (int r = 0; r < 8; ++r) {
        const size_t m = (size_t)(mBase + wm * 16 + r + 8 * h);
#pragma unroll
        for (int st = 0; st < 4; ++st)
            __builtin_nontemporal_store(acc[st][r],
                &logit[m * VOCAB + (n0 + st * 16 + ml)]);
    }
}

// ---------------------------------------------------------------------------
// Kernel 3: logit_w[row,k] = 2 * P_k / Z - 1
// ---------------------------------------------------------------------------
__global__ __launch_bounds__(256) void softmax_proj_kernel(
    const float* __restrict__ logit,
    float* __restrict__ out)
{
    const int row = blockIdx.x;
    const int tid = threadIdx.x;
    const float* L = logit + (size_t)row * VOCAB;

    __shared__ float sred[256];
    __shared__ float sres[16];

    // pass 1: row max
    float m = -3.402823466e38f;
    for (int v = tid; v < VOCAB; v += 256)
        m = fmaxf(m, L[v]);
    sred[tid] = m;
    __syncthreads();
    for (int s = 128; s > 0; s >>= 1) {
        if (tid < s) sred[tid] = fmaxf(sred[tid], sred[tid + s]);
        __syncthreads();
    }
    const float rowmax = sred[0];
    __syncthreads();

    // pass 2: Z and per-bit partial sums
    float Z = 0.f;
    float P[BITS];
#pragma unroll
    for (int k = 0; k < BITS; ++k) P[k] = 0.f;
    for (int v = tid; v < VOCAB; v += 256) {
        float e = __expf(L[v] - rowmax);
        Z += e;
#pragma unroll
        for (int k = 0; k < BITS; ++k)
            if ((v >> (BITS - 1 - k)) & 1) P[k] += e;
    }

    float vals[16];
    vals[0] = Z;
#pragma unroll
    for (int k = 0; k < BITS; ++k) vals[k + 1] = P[k];

    for (int q = 0; q < 16; ++q) {
        sred[tid] = vals[q];
        __syncthreads();
        for (int s = 128; s > 0; s >>= 1) {
            if (tid < s) sred[tid] += sred[tid + s];
            __syncthreads();
        }
        if (tid == 0) sres[q] = sred[0];
        __syncthreads();
    }

    if (tid < BITS)
        out[(size_t)row * BITS + tid] = 2.f * sres[tid + 1] / sres[0] - 1.f;
}

// ---------------------------------------------------------------------------
// Launch
// ---------------------------------------------------------------------------
extern "C" void kernel_launch(void* const* d_in, const int* in_sizes, int n_in,
                              void* d_out, int out_size, void* d_ws, size_t ws_size,
                              hipStream_t stream) {
    (void)in_sizes; (void)n_in; (void)out_size; (void)ws_size;

    const int*   ids        = (const int*)d_in[0];
    const float* tensor     = (const float*)d_in[1];
    const float* weight     = (const float*)d_in[2];
    const float* weight_bit = (const float*)d_in[3];

    float* out     = (float*)d_out;
    float* id_emb  = out + OUT_ID_EMB;
    float* bit_emb = out + OUT_BIT_EMB;
    float* logit   = out + OUT_LOGIT;
    float* logit_w = out + OUT_LOGIT_W;

    unsigned char* ws = (unsigned char*)d_ws;
    unsigned short* a_hi = (unsigned short*)(ws + WS_A_HI);
    unsigned short* a_lo = (unsigned short*)(ws + WS_A_LO);
    unsigned short* w_hi = (unsigned short*)(ws + WS_W_HI);
    unsigned short* w_lo = (unsigned short*)(ws + WS_W_LO);

    // 0) one-shot fp32 -> bf16 hi/lo plane splits (bandwidth-bound)
    split_planes_kernel<<<dim3((NTOK * DIM / 4 + 255) / 256), dim3(256), 0, stream>>>(
        tensor, a_hi, a_lo, NTOK * DIM / 4);
    split_planes_kernel<<<dim3((VOCAB * DIM / 4 + 255) / 256), dim3(256), 0, stream>>>(
        weight, w_hi, w_lo, VOCAB * DIM / 4);

    // 1) gather + bit embedding (bandwidth-bound, LDS-staged weight_bit)
    embed_kernel<<<dim3(NTOK), dim3(256), 0, stream>>>(
        ids, weight, weight_bit, id_emb, bit_emb);

    // 2) big GEMM on the WMMA pipe (bf16 split precision, async LDS staging)
    gemm_logit_kernel<<<dim3(VOCAB / 128, NTOK / 64), dim3(256), 0, stream>>>(
        a_hi, a_lo, w_hi, w_lo, logit);

    // 3) softmax projection onto bit codes
    softmax_proj_kernel<<<dim3(NTOK), dim3(256), 0, stream>>>(logit, logit_w);
}
